// MaximumMeanDiscrepancy_49984829391268
// MI455X (gfx1250) — matrix operand order repaired and checked by reading, not compiled, not run on hardware
//
#include <hip/hip_runtime.h>
#include <hip/hip_bf16.h>

typedef __attribute__((ext_vector_type(16))) __bf16 v16bf;
typedef __attribute__((ext_vector_type(8)))  __bf16 v8bf;
typedef __attribute__((ext_vector_type(4)))  __bf16 v4bf;
typedef __attribute__((ext_vector_type(8)))  float  v8f;

#define B_ROWS 4096
#define N_ROWS 8192
#define D_COLS 256

__device__ __forceinline__ const float* row_ptr(const float* __restrict__ S,
                                                const float* __restrict__ T, int i) {
  return (i < B_ROWS) ? (S + (size_t)i * D_COLS) : (T + (size_t)(i - B_ROWS) * D_COLS);
}

// ---------------------------------------------------------------------------
// FAST-PATH Kernel 1: per-row sq norms + f32 -> (bf16 hi, bf16 lo) planes.
// One wave per row; lane owns 8 elements (two float4 runs).
// ---------------------------------------------------------------------------
__global__ __launch_bounds__(256) void mmd_prep_kernel(const float* __restrict__ S,
                                                       const float* __restrict__ T,
                                                       __bf16* __restrict__ hi,
                                                       __bf16* __restrict__ lo,
                                                       float* __restrict__ sq) {
  const int lane = threadIdx.x & 31;
  const int wave = threadIdx.x >> 5;
  const int row  = blockIdx.x * 8 + wave;
  const float4* r4 = (const float4*)row_ptr(S, T, row);
  const float4 a = r4[lane];        // floats [4*lane .. 4*lane+3]
  const float4 b = r4[lane + 32];   // floats [128+4*lane ..]

  float xs[4] = {a.x, a.y, a.z, a.w};
  float ys[4] = {b.x, b.y, b.z, b.w};
  v4bf ah, al, bh, bl;
  float s = 0.0f;
  #pragma unroll
  for (int e = 0; e < 4; ++e) {
    const float x = xs[e];
    const __bf16 hx = (__bf16)x;
    ah[e] = hx;
    al[e] = (__bf16)(x - (float)hx);
    const float y = ys[e];
    const __bf16 hy = (__bf16)y;
    bh[e] = hy;
    bl[e] = (__bf16)(y - (float)hy);
    s = fmaf(x, x, s);
    s = fmaf(y, y, s);
  }
  __bf16* hrow = hi + (size_t)row * D_COLS;
  __bf16* lrow = lo + (size_t)row * D_COLS;
  *(v4bf*)(hrow + 4 * lane)       = ah;
  *(v4bf*)(hrow + 128 + 4 * lane) = bh;
  *(v4bf*)(lrow + 4 * lane)       = al;
  *(v4bf*)(lrow + 128 + 4 * lane) = bl;

  #pragma unroll
  for (int m = 16; m >= 1; m >>= 1) s += __shfl_xor(s, m, 32);
  if (lane == 0) sq[row] = s;
}

// ---------------------------------------------------------------------------
// FALLBACK Kernel 1: per-row squared norms only.
// ---------------------------------------------------------------------------
__global__ __launch_bounds__(256) void mmd_sq_kernel(const float* __restrict__ S,
                                                     const float* __restrict__ T,
                                                     float* __restrict__ sq) {
  const int lane = threadIdx.x & 31;
  const int wave = threadIdx.x >> 5;
  const int row  = blockIdx.x * 8 + wave;
  const float4* r4 = (const float4*)row_ptr(S, T, row);
  float4 a = r4[lane];
  float4 b = r4[lane + 32];
  float s = a.x * a.x;
  s = fmaf(a.y, a.y, s); s = fmaf(a.z, a.z, s); s = fmaf(a.w, a.w, s);
  s = fmaf(b.x, b.x, s); s = fmaf(b.y, b.y, s);
  s = fmaf(b.z, b.z, s); s = fmaf(b.w, b.w, s);
  #pragma unroll
  for (int m = 16; m >= 1; m >>= 1) s += __shfl_xor(s, m, 32);
  if (lane == 0) sq[row] = s;
}

// ---------------------------------------------------------------------------
// Kernel 2: bandwidth from sum(L2) = 2*N*sum(sq) - 2*||colsum||^2.
// One block; writes gam[i] = -log2(e) / (bw * 2^i).
// ---------------------------------------------------------------------------
__global__ __launch_bounds__(256) void mmd_bw_kernel(const float* __restrict__ S,
                                                     const float* __restrict__ T,
                                                     const float* __restrict__ sq,
                                                     float* __restrict__ gam) {
  const int d = threadIdx.x;
  float cs = 0.0f;
  for (int i = 0; i < B_ROWS; ++i) cs += S[(size_t)i * D_COLS + d];
  for (int i = 0; i < B_ROWS; ++i) cs += T[(size_t)i * D_COLS + d];
  float sqp = 0.0f;
  for (int k = 0; k < N_ROWS / 256; ++k) sqp += sq[d + 256 * k];

  __shared__ float r1[256];
  __shared__ float r2[256];
  r1[d] = cs * cs;
  r2[d] = sqp;
  __syncthreads();
  #pragma unroll
  for (int s = 128; s > 0; s >>= 1) {
    if (d < s) { r1[d] += r1[d + s]; r2[d] += r2[d + s]; }
    __syncthreads();
  }
  if (d == 0) {
    const float sumsq = r2[0];
    const float colsq = r1[0];
    const float sumL2 = 2.0f * (float)N_ROWS * sumsq - 2.0f * colsq;
    float bw = sumL2 / (float)((double)N_ROWS * (double)N_ROWS - (double)N_ROWS);
    bw = bw / 4.0f;  // KERNEL_MUL ** (KERNEL_NUM // 2)
    const float log2e = 1.4426950408889634f;
    #pragma unroll
    for (int i = 0; i < 5; ++i) gam[i] = -log2e / (bw * (float)(1 << i));
  }
}

// ---------------------------------------------------------------------------
// FAST-PATH Kernel 3: wave computes a 16x32 strip (two 16x16 tiles sharing
// the A hi/lo fragments). Block = 8 waves covering 64x64. All operands are
// pre-split bf16 -> zero conversion VALU in the hot loop; 6 WMMA per A-load.
// ---------------------------------------------------------------------------
__global__ __launch_bounds__(256) void mmd_main2_kernel(const __bf16* __restrict__ hi,
                                                        const __bf16* __restrict__ lo,
                                                        const float* __restrict__ sq,
                                                        const float* __restrict__ gam,
                                                        float* __restrict__ partials) {
  const int tid  = threadIdx.x;
  const int lane = tid & 31;
  const int wave = tid >> 5;

  const int tileI = blockIdx.y * 64 + (wave & 3) * 16;   // row tile (M)
  const int tileJ = blockIdx.x * 64 + (wave >> 2) * 32;  // first col tile (N)

  // A (16x32 bf16): lane -> M=lane%16, K = koff .. +15 contiguous, koff=(lane/16)*16
  const int mA    = tileI + (lane & 15);
  const int koffA = (lane >> 4) * 16;
  // B (32x16): lane -> N=lane%16, K = (lane/16)*16 .. +15 contiguous
  const int nB    = tileJ + (lane & 15);
  const int kblkB = (lane >> 4) * 16;

  const __bf16* hA  = hi + (size_t)mA * D_COLS;
  const __bf16* lA  = lo + (size_t)mA * D_COLS;
  const __bf16* hB0 = hi + (size_t)nB * D_COLS;
  const __bf16* lB0 = lo + (size_t)nB * D_COLS;
  const __bf16* hB1 = hB0 + (size_t)16 * D_COLS;
  const __bf16* lB1 = lB0 + (size_t)16 * D_COLS;

  v8f acc0 = {};
  v8f acc1 = {};
  #pragma unroll 2
  for (int kb = 0; kb < D_COLS; kb += 32) {
    v16bf ahi, alo;
    *(v8bf*)&ahi         = *(const v8bf*)(hA + kb + koffA);
    *(((v8bf*)&ahi) + 1) = *(const v8bf*)(hA + kb + 8 + koffA);
    *(v8bf*)&alo         = *(const v8bf*)(lA + kb + koffA);
    *(((v8bf*)&alo) + 1) = *(const v8bf*)(lA + kb + 8 + koffA);
    const v16bf b0h = *(const v16bf*)(hB0 + kb + kblkB);
    const v16bf b0l = *(const v16bf*)(lB0 + kb + kblkB);
    const v16bf b1h = *(const v16bf*)(hB1 + kb + kblkB);
    const v16bf b1l = *(const v16bf*)(lB1 + kb + kblkB);

    acc0 = __builtin_amdgcn_wmma_f32_16x16x32_bf16(false, alo, false, b0h,
                                                   (short)0, acc0, false, false);
    acc0 = __builtin_amdgcn_wmma_f32_16x16x32_bf16(false, ahi, false, b0l,
                                                   (short)0, acc0, false, false);
    acc0 = __builtin_amdgcn_wmma_f32_16x16x32_bf16(false, ahi, false, b0h,
                                                   (short)0, acc0, false, false);
    acc1 = __builtin_amdgcn_wmma_f32_16x16x32_bf16(false, alo, false, b1h,
                                                   (short)0, acc1, false, false);
    acc1 = __builtin_amdgcn_wmma_f32_16x16x32_bf16(false, ahi, false, b1l,
                                                   (short)0, acc1, false, false);
    acc1 = __builtin_amdgcn_wmma_f32_16x16x32_bf16(false, ahi, false, b1h,
                                                   (short)0, acc1, false, false);
  }

  // Epilogue. C/D layout: lane holds N = lane%16, rows M = (lane/16)*8 + r.
  const float g0 = gam[0], g1 = gam[1], g2 = gam[2], g3 = gam[3], g4 = gam[4];
  const float sqN0 = sq[nB];
  const float sqN1 = sq[nB + 16];
  const float4* sp = (const float4*)(sq + tileI + (lane >> 4) * 8);
  const float4 s0 = sp[0], s1 = sp[1];
  const float sqM[8] = {s0.x, s0.y, s0.z, s0.w, s1.x, s1.y, s1.z, s1.w};

  float part = 0.0f;
  #pragma unroll
  for (int r = 0; r < 8; ++r) {
    float l2a = fmaxf(sqM[r] + sqN0 - 2.0f * acc0[r], 0.0f);
    float l2b = fmaxf(sqM[r] + sqN1 - 2.0f * acc1[r], 0.0f);
    part += __builtin_amdgcn_exp2f(l2a * g0) + __builtin_amdgcn_exp2f(l2b * g0);
    part += __builtin_amdgcn_exp2f(l2a * g1) + __builtin_amdgcn_exp2f(l2b * g1);
    part += __builtin_amdgcn_exp2f(l2a * g2) + __builtin_amdgcn_exp2f(l2b * g2);
    part += __builtin_amdgcn_exp2f(l2a * g3) + __builtin_amdgcn_exp2f(l2b * g3);
    part += __builtin_amdgcn_exp2f(l2a * g4) + __builtin_amdgcn_exp2f(l2b * g4);
  }
  // Block weight: +1 same half, -1 cross half (both tiles lie in one half:
  // the 64-wide block never straddles the 4096 boundary).
  const float w = ((tileI < B_ROWS) == (tileJ < B_ROWS)) ? 1.0f : -1.0f;
  part *= w;

  __shared__ float red[256];
  red[tid] = part;
  __syncthreads();
  #pragma unroll
  for (int s = 128; s > 0; s >>= 1) {
    if (tid < s) red[tid] += red[tid + s];
    __syncthreads();
  }
  if (tid == 0) partials[blockIdx.y * gridDim.x + blockIdx.x] = red[0];
}

// ---------------------------------------------------------------------------
// FALLBACK Kernel 3: on-the-fly hi/lo split (round-1 version); needs only
// ~170 KB of workspace.
// ---------------------------------------------------------------------------
__global__ __launch_bounds__(256) void mmd_main_kernel(const float* __restrict__ S,
                                                       const float* __restrict__ T,
                                                       const float* __restrict__ sq,
                                                       const float* __restrict__ gam,
                                                       float* __restrict__ partials) {
  const int tid  = threadIdx.x;
  const int lane = tid & 31;
  const int wave = tid >> 5;
  const int tileI = blockIdx.y * 64 + (wave & 3) * 16;
  const int tileJ = blockIdx.x * 32 + (wave >> 2) * 16;
  const int mA    = tileI + (lane & 15);
  const int koffA = (lane >> 4) * 16;
  const int nB    = tileJ + (lane & 15);
  const int kblkB = (lane >> 4) * 16;
  const float* rowA = row_ptr(S, T, mA);
  const float* rowB = row_ptr(S, T, nB);

  v8f acc = {};
  for (int kb = 0; kb < D_COLS; kb += 32) {
    float av[16], bv[16];
    {
      const float4 a0 = *(const float4*)(rowA + kb + koffA + 0);
      const float4 a1 = *(const float4*)(rowA + kb + koffA + 4);
      const float4 a2 = *(const float4*)(rowA + kb + koffA + 8);
      const float4 a3 = *(const float4*)(rowA + kb + koffA + 12);
      av[0]=a0.x; av[1]=a0.y; av[2]=a0.z; av[3]=a0.w;
      av[4]=a1.x; av[5]=a1.y; av[6]=a1.z; av[7]=a1.w;
      av[8]=a2.x; av[9]=a2.y; av[10]=a2.z; av[11]=a2.w;
      av[12]=a3.x; av[13]=a3.y; av[14]=a3.z; av[15]=a3.w;
      const float4 b0 = *(const float4*)(rowB + kb + kblkB + 0);
      const float4 b1 = *(const float4*)(rowB + kb + kblkB + 4);
      const float4 b2 = *(const float4*)(rowB + kb + kblkB + 8);
      const float4 b3 = *(const float4*)(rowB + kb + kblkB + 12);
      bv[0]=b0.x; bv[1]=b0.y; bv[2]=b0.z; bv[3]=b0.w;
      bv[4]=b1.x; bv[5]=b1.y; bv[6]=b1.z; bv[7]=b1.w;
      bv[8]=b2.x; bv[9]=b2.y; bv[10]=b2.z; bv[11]=b2.w;
      bv[12]=b3.x; bv[13]=b3.y; bv[14]=b3.z; bv[15]=b3.w;
    }
    v16bf ahi, alo, bhi, blo;
    #pragma unroll
    for (int e = 0; e < 16; ++e) {
      const float x = av[e];
      const __bf16 hx = (__bf16)x;
      ahi[e] = hx;
      alo[e] = (__bf16)(x - (float)hx);
      const float y = bv[e];
      const __bf16 hy = (__bf16)y;
      bhi[e] = hy;
      blo[e] = (__bf16)(y - (float)hy);
    }
    acc = __builtin_amdgcn_wmma_f32_16x16x32_bf16(false, alo, false, bhi,
                                                  (short)0, acc, false, false);
    acc = __builtin_amdgcn_wmma_f32_16x16x32_bf16(false, ahi, false, blo,
                                                  (short)0, acc, false, false);
    acc = __builtin_amdgcn_wmma_f32_16x16x32_bf16(false, ahi, false, bhi,
                                                  (short)0, acc, false, false);
  }

  const float g0 = gam[0], g1 = gam[1], g2 = gam[2], g3 = gam[3], g4 = gam[4];
  const float sqN = sq[nB];
  const float4* sp = (const float4*)(sq + tileI + (lane >> 4) * 8);
  const float4 s0 = sp[0], s1 = sp[1];
  const float sqM[8] = {s0.x, s0.y, s0.z, s0.w, s1.x, s1.y, s1.z, s1.w};

  float part = 0.0f;
  #pragma unroll
  for (int r = 0; r < 8; ++r) {
    float l2 = fmaxf(sqM[r] + sqN - 2.0f * acc[r], 0.0f);
    part += __builtin_amdgcn_exp2f(l2 * g0);
    part += __builtin_amdgcn_exp2f(l2 * g1);
    part += __builtin_amdgcn_exp2f(l2 * g2);
    part += __builtin_amdgcn_exp2f(l2 * g3);
    part += __builtin_amdgcn_exp2f(l2 * g4);
  }
  const float w = ((tileI < B_ROWS) == (tileJ < B_ROWS)) ? 1.0f : -1.0f;
  part *= w;

  __shared__ float red[256];
  red[tid] = part;
  __syncthreads();
  #pragma unroll
  for (int s = 128; s > 0; s >>= 1) {
    if (tid < s) red[tid] += red[tid + s];
    __syncthreads();
  }
  if (tid == 0) partials[blockIdx.y * gridDim.x + blockIdx.x] = red[0];
}

// ---------------------------------------------------------------------------
// Kernel 4: deterministic final reduction of block partials.
// ---------------------------------------------------------------------------
__global__ __launch_bounds__(256) void mmd_finish_kernel(const float* __restrict__ partials,
                                                         int n,
                                                         float* __restrict__ out) {
  const int tid = threadIdx.x;
  float s = 0.0f;
  for (int k = tid; k < n; k += 256) s += partials[k];
  __shared__ float red[256];
  red[tid] = s;
  __syncthreads();
  #pragma unroll
  for (int st = 128; st > 0; st >>= 1) {
    if (tid < st) red[tid] += red[tid + st];
    __syncthreads();
  }
  if (tid == 0) out[0] = red[0] * (1.0f / ((float)B_ROWS * (float)B_ROWS));
}

extern "C" void kernel_launch(void* const* d_in, const int* in_sizes, int n_in,
                              void* d_out, int out_size, void* d_ws, size_t ws_size,
                              hipStream_t stream) {
  const float* S = (const float*)d_in[0];  // source_features [4096,256] f32
  const float* T = (const float*)d_in[1];  // target_features [4096,256] f32

  const size_t PLANE = (size_t)N_ROWS * D_COLS;  // 2M elements
  // Fast-path workspace need: 2 bf16 planes + sq + gam + 128*128 partials.
  const size_t need_fast = PLANE * 2 * sizeof(__bf16)
                         + (size_t)(N_ROWS + 8 + 128 * 128) * sizeof(float);

  if (ws_size >= need_fast) {
    __bf16* hi = (__bf16*)d_ws;
    __bf16* lo = hi + PLANE;
    float* sq       = (float*)(lo + PLANE);
    float* gam      = sq + N_ROWS;
    float* partials = gam + 8;

    mmd_prep_kernel<<<N_ROWS / 8, 256, 0, stream>>>(S, T, hi, lo, sq);
    mmd_bw_kernel<<<1, 256, 0, stream>>>(S, T, sq, gam);
    dim3 grid(N_ROWS / 64, N_ROWS / 64);  // (128, 128)
    mmd_main2_kernel<<<grid, 256, 0, stream>>>(hi, lo, sq, gam, partials);
    mmd_finish_kernel<<<1, 256, 0, stream>>>(partials, 128 * 128, (float*)d_out);
  } else {
    float* wsf = (float*)d_ws;
    float* sq       = wsf;
    float* gam      = wsf + N_ROWS;
    float* partials = gam + 8;

    mmd_sq_kernel<<<N_ROWS / 8, 256, 0, stream>>>(S, T, sq);
    mmd_bw_kernel<<<1, 256, 0, stream>>>(S, T, sq, gam);
    dim3 grid(N_ROWS / 32, N_ROWS / 64);  // (256, 128)
    mmd_main_kernel<<<grid, 256, 0, stream>>>(S, T, sq, gam, partials);
    mmd_finish_kernel<<<1, 256, 0, stream>>>(partials, 256 * 128, (float*)d_out);
  }
}
